// DeepFFM_58574763983376
// MI455X (gfx1250) — matrix operand order repaired and checked by reading, not compiled, not run on hardware
//
#include <hip/hip_runtime.h>
#include <math.h>

// ---------------- problem constants ----------------
#define FIELD 39
#define NFEAT 50000
#define NGRP  8
#define EMB   8
#define D0    312   // FIELD*EMB
#define K0P   320   // D0 padded to multiple of 32
#define N1    512
#define N2    256
#define N3    128
#define MROWS 128   // batch rows per block

// LDS map (ushort element offsets). Lifetime-based aliasing:
//   H0 (gather in)   [0,      40960)  80KB   dead after L1
//   H1 (L1 out)      [40960, 106496) 128KB   dead after L2
//   WP (weight panel)[106496,147456)  80KB   reused every stage
//   H2 (L2 out)      alias H0 (64KB <= 80KB)
//   H3 (L3 out)      alias H1 (32KB <= 128KB)
//   BASE (float[128])[147456,147712)  512B   live whole kernel
#define H0_OFS   0
#define H1_OFS   40960
#define WP_OFS   106496
#define H2_OFS   0
#define H3_OFS   40960
#define BASE_OFS 147456
#define SMEM_USH 147712   // 295424 bytes (< 320KB WG limit)

typedef __bf16 v16bf __attribute__((ext_vector_type(16)));
typedef float  v8f   __attribute__((ext_vector_type(8)));

union Frag16 { v16bf v; uint4 u[2]; };

__device__ __forceinline__ unsigned short f2bf(float f) {
  unsigned int u = __float_as_uint(f);
  u += 0x7FFFu + ((u >> 16) & 1u);          // round-to-nearest-even
  return (unsigned short)(u >> 16);
}
__device__ __forceinline__ float bf2f(unsigned short h) {
  return __uint_as_float(((unsigned int)h) << 16);
}

// ---------------- S = masked field-pair interaction matrix (batch-invariant) ----------------
__global__ void ffm_smask_kernel(const float* __restrict__ FM_V, float* __restrict__ Sm) {
  int t = blockIdx.x * blockDim.x + threadIdx.x;
  if (t >= FIELD * FIELD) return;
  int i = t / FIELD, j = t % FIELD;
  float s = 0.f;
  if (j > i) {  // strict upper triangle (mask k=1)
    const float* vi = FM_V + ((size_t)i * NFEAT + (size_t)i * 1000) * (NGRP * EMB) + (j % NGRP) * EMB;
    const float* vj = FM_V + ((size_t)j * NFEAT + (size_t)j * 1000) * (NGRP * EMB) + (i % NGRP) * EMB;
#pragma unroll
    for (int e = 0; e < EMB; ++e) s += vi[e] * vj[e];
  }
  Sm[t] = s;
}

// ---------------- W [K,N] f32 -> Wt [N,Kpad] bf16 (transposed, K zero-padded) ----------------
__global__ void convert_wt_kernel(const float* __restrict__ W, unsigned short* __restrict__ Wt,
                                  int K, int Kpad, int N) {
  int t = blockIdx.x * blockDim.x + threadIdx.x;
  if (t >= N * Kpad) return;
  int n = t / Kpad, k = t % Kpad;
  Wt[t] = (k < K) ? f2bf(W[(size_t)k * N + n]) : (unsigned short)0;
}

// ---------------- fused kernel helpers ----------------
// 16x128 strip of WMMA tiles: A rows (wave*16..+16) x 8 n-tiles, K-loop in steps of 32.
// bf16 A frag: lane ln=M, half h -> K 16h..16h+15. B frag: lane ln=N, half h -> K 16h..16h+15.
__device__ __forceinline__ void mm_tiles(unsigned short* sm, int aOfs, int aStride,
                                         int bOfs, int bStride, int kLen,
                                         int wave, int ln, int h, v8f acc[8]) {
  for (int k0 = 0; k0 < kLen; k0 += 32) {
    Frag16 af;
    const int am = aOfs + (wave * 16 + ln) * aStride + k0;
    af.u[0] = *(const uint4*)&sm[am + h * 16];
    af.u[1] = *(const uint4*)&sm[am + h * 16 + 8];
#pragma unroll
    for (int nt = 0; nt < 8; ++nt) {
      Frag16 bf;
      const int bo = bOfs + (nt * 16 + ln) * bStride + k0 + h * 16;
      bf.u[0] = *(const uint4*)&sm[bo];
      bf.u[1] = *(const uint4*)&sm[bo + 8];
      acc[nt] = __builtin_amdgcn_wmma_f32_16x16x32_bf16(
          false, af.v, false, bf.v, (short)0, acc[nt], false, false);
    }
  }
}

// C/D layout: VGPR r -> M = r + 8h, lane ln -> N. ReLU + bf16 back into LDS.
__device__ __forceinline__ void epilogue(unsigned short* sm, int dOfs, int dStride,
                                         int colBase, const float* __restrict__ bias,
                                         int wave, int ln, int h, v8f acc[8]) {
  const int r0 = wave * 16 + h * 8;
#pragma unroll
  for (int nt = 0; nt < 8; ++nt) {
    const int col = colBase + nt * 16 + ln;
    const float bv = bias[col];
#pragma unroll
    for (int r = 0; r < 8; ++r) {
      float v = acc[nt][r] + bv;
      v = v > 0.f ? v : 0.f;
      sm[dOfs + (r0 + r) * dStride + col] = f2bf(v);
    }
  }
}

__device__ __forceinline__ void zacc(v8f acc[8]) {
#pragma unroll
  for (int nt = 0; nt < 8; ++nt) {
    v8f z = {0.f, 0.f, 0.f, 0.f, 0.f, 0.f, 0.f, 0.f};
    acc[nt] = z;
  }
}

// ---------------- fully fused DeepFFM forward: one block = 128 samples ----------------
// All activations LDS-resident; only ids/vals/weights (L2-hot) in, 128 sigmoids out.
__launch_bounds__(256)
__global__ void deepffm_fused_kernel(
    const int*   __restrict__ feat_ids, const float* __restrict__ feat_vals,
    const float* __restrict__ FM_W,     const float* __restrict__ FM_B,
    const float* __restrict__ Sm,       const float* __restrict__ embedding,
    const unsigned short* __restrict__ Wt0, const float* __restrict__ B0,
    const unsigned short* __restrict__ Wt1, const float* __restrict__ B1,
    const unsigned short* __restrict__ Wt2, const float* __restrict__ B2,
    const float* __restrict__ outW, const float* __restrict__ outB,
    float* __restrict__ out) {
  __shared__ __align__(16) unsigned short sm[SMEM_USH];
  float* sBase = (float*)&sm[BASE_OFS];

  const int tid  = threadIdx.x;
  const int wave = tid >> 5;
  const int lane = tid & 31;
  const int ln   = lane & 15;
  const int h    = lane >> 4;
  const int bm   = blockIdx.x * MROWS;

  // ---- phase G: FM linear+interaction base (threads 0..127), embedding gather (all) ----
  if (tid < MROWS) {
    const int b = bm + tid;
    const int*   idp = feat_ids  + (size_t)b * FIELD;
    const float* vp  = feat_vals + (size_t)b * FIELD;
    float vv[FIELD];
    float lin = FM_B[0];
#pragma unroll
    for (int i = 0; i < FIELD; ++i) {
      float v = vp[i];
      vv[i] = v;
      lin += FM_W[idp[i]] * v;
    }
    float inter = 0.f;
    for (int i = 0; i < FIELD - 1; ++i) {
      float a = 0.f;
      for (int j = i + 1; j < FIELD; ++j) a += Sm[i * FIELD + j] * vv[j];
      inter += vv[i] * a;
    }
    sBase[tid] = lin + inter;
  }
  for (int idx = tid; idx < MROWS * K0P; idx += 256) {
    const int m = idx / K0P, k = idx % K0P;
    unsigned short r = 0;
    if (k < D0) {
      const int f = k >> 3, e = k & 7;
      const int id = feat_ids[(size_t)(bm + m) * FIELD + f];
      r = f2bf(embedding[(size_t)id * EMB + e]);
    }
    sm[H0_OFS + idx] = r;
  }
  __syncthreads();

  // ---- layer 1: H1[128,512] = relu(H0[128,320] x W0 + B0), 4 col-groups of 128 ----
  for (int g = 0; g < 4; ++g) {
    {  // stage contiguous 80KB weight panel (Wt rows g*128..+127, full K)
      const uint4* src = (const uint4*)(Wt0 + (size_t)g * 128 * K0P);
      uint4* dst = (uint4*)&sm[WP_OFS];
      for (int i = tid; i < 128 * K0P / 8; i += 256) dst[i] = src[i];
      if (g < 3) __builtin_prefetch(Wt0 + (size_t)(g + 1) * 128 * K0P + tid * 128, 0, 1);
    }
    __syncthreads();
    v8f acc[8];
    zacc(acc);
    mm_tiles(sm, H0_OFS, K0P, WP_OFS, K0P, K0P, wave, ln, h, acc);
    epilogue(sm, H1_OFS, N1, g * 128, B0, wave, ln, h, acc);
    __syncthreads();
  }

  // ---- layer 2: H2[128,256] = relu(H1 x W1 + B1), 2 col-groups, K split in 2 panels ----
  for (int g = 0; g < 2; ++g) {
    v8f acc[8];
    zacc(acc);
    for (int kp = 0; kp < 2; ++kp) {
      {  // stage [128 n][256 k] half-panel (Wt1 row stride 512)
        uint4* dst = (uint4*)&sm[WP_OFS];
        for (int i = tid; i < 128 * 32; i += 256) {
          const int n = i >> 5, c = i & 31;
          dst[i] = *(const uint4*)(Wt1 + (size_t)(g * 128 + n) * N1 + kp * 256 + c * 8);
        }
      }
      __syncthreads();
      mm_tiles(sm, H1_OFS + kp * 256, N1, WP_OFS, 256, 256, wave, ln, h, acc);
      __syncthreads();
    }
    epilogue(sm, H2_OFS, N2, g * 128, B1, wave, ln, h, acc);
  }

  // ---- layer 3: H3[128,128] = relu(H2 x W2 + B2), single group, 64KB panel ----
  {
    const uint4* src = (const uint4*)Wt2;
    uint4* dst = (uint4*)&sm[WP_OFS];
    for (int i = tid; i < 128 * N2 / 8; i += 256) dst[i] = src[i];
  }
  __syncthreads();   // also orders layer-2 epilogue writes (H2) before reads
  {
    v8f acc[8];
    zacc(acc);
    mm_tiles(sm, H2_OFS, N2, WP_OFS, N2, N2, wave, ln, h, acc);
    epilogue(sm, H3_OFS, N3, 0, B2, wave, ln, h, acc);
  }
  __syncthreads();

  // ---- final: out = sigmoid(base + H3 . outW + outB) ----
  if (tid < MROWS) {
    const uint4* hp = (const uint4*)&sm[H3_OFS + tid * N3];
    float s = 0.f;
#pragma unroll
    for (int c = 0; c < N3 / 8; ++c) {
      uint4 u = hp[c];
      const unsigned short* e = (const unsigned short*)&u;
#pragma unroll
      for (int q = 0; q < 8; ++q) s += bf2f(e[q]) * outW[c * 8 + q];
    }
    const float x = sBase[tid] + s + outB[0];
    out[bm + tid] = 1.f / (1.f + __expf(-x));
  }
}

// ---------------- launcher ----------------
extern "C" void kernel_launch(void* const* d_in, const int* in_sizes, int n_in,
                              void* d_out, int out_size, void* d_ws, size_t ws_size,
                              hipStream_t stream) {
  const int*   feat_ids  = (const int*)  d_in[0];
  const float* feat_vals = (const float*)d_in[1];
  const float* FM_W      = (const float*)d_in[2];
  const float* FM_V      = (const float*)d_in[3];
  const float* FM_B      = (const float*)d_in[4];
  const float* embedding = (const float*)d_in[5];
  const float* deepW0    = (const float*)d_in[6];
  const float* deepB0    = (const float*)d_in[7];
  const float* deepW1    = (const float*)d_in[8];
  const float* deepB1    = (const float*)d_in[9];
  const float* deepW2    = (const float*)d_in[10];
  const float* deepB2    = (const float*)d_in[11];
  const float* outW      = (const float*)d_in[12];
  const float* outB      = (const float*)d_in[13];
  float* out = (float*)d_out;
  const int B = in_sizes[0] / FIELD;   // 16384 (multiple of 128)

  char* ws = (char*)d_ws;
  size_t off = 0;
  auto wsalloc = [&](size_t bytes) -> char* {
    char* p = ws + off;
    off += (bytes + 255) & ~(size_t)255;
    return p;
  };
  float*          Sm  = (float*)         wsalloc((size_t)FIELD * FIELD * 4);
  unsigned short* Wt0 = (unsigned short*)wsalloc((size_t)N1 * K0P * 2);
  unsigned short* Wt1 = (unsigned short*)wsalloc((size_t)N2 * N1 * 2);
  unsigned short* Wt2 = (unsigned short*)wsalloc((size_t)N3 * N2 * 2);
  (void)ws_size; (void)n_in; (void)out_size;

  ffm_smask_kernel<<<(FIELD * FIELD + 255) / 256, 256, 0, stream>>>(FM_V, Sm);
  convert_wt_kernel<<<(N1 * K0P + 255) / 256, 256, 0, stream>>>(deepW0, Wt0, D0, K0P, N1);
  convert_wt_kernel<<<(N2 * N1 + 255) / 256, 256, 0, stream>>>(deepW1, Wt1, N1, N1, N2);
  convert_wt_kernel<<<(N3 * N2 + 255) / 256, 256, 0, stream>>>(deepW2, Wt2, N2, N2, N3);

  deepffm_fused_kernel<<<B / MROWS, 256, 0, stream>>>(
      feat_ids, feat_vals, FM_W, FM_B, Sm, embedding,
      Wt0, deepB0, Wt1, deepB1, Wt2, deepB2, outW, outB, out);
}
